// TensorNetInteraction_68058051772520
// MI455X (gfx1250) — matrix operand order repaired and checked by reading, not compiled, not run on hardware
//
#include <hip/hip_runtime.h>
#include <hip/hip_bf16.h>

#define N_ATOMS 25000
#define N_TILES 1563            // ceil(25000/16)
#define N_PAD   (N_TILES * 16)  // 25008
#define FDIM    64
#define RBFDIM  32
#define EDGES   400000

typedef __bf16 bf16_t;
typedef bf16_t v16bf __attribute__((ext_vector_type(16)));
typedef float  v8f   __attribute__((ext_vector_type(8)));
typedef float  v2f   __attribute__((ext_vector_type(2)));

// 16-bit A-matrix 16x32 VGPR layout: VGPR r holds K = k0,k0+1 for lane-half g
__device__ __forceinline__ int kbase(int r, int g) {
    return 2 * (r & 3) + 8 * g + 16 * (r >> 2);
}
// inverse mapping: K (0..31) -> (g, r, pair) within A fragment
__device__ __forceinline__ void kinv(int k, int& g, int& r, int& pair) {
    g    = (k >> 3) & 1;
    r    = ((k & 7) >> 1) + 4 * (k >> 4);
    pair = k & 1;
}

__device__ __forceinline__ float silu_f(float x) { return x / (1.0f + __expf(-x)); }

__device__ __forceinline__ v8f wmma_bf16(v16bf a, v16bf b, v8f c) {
    return __builtin_amdgcn_wmma_f32_16x16x32_bf16(false, a, false, b, (short)0, c, false, false);
}

// --------------------------------------------------------------------------
// Weight pre-swizzle: W[OUT][IN] f32 -> bf16 B-fragment order
// dst[((kt*nnt + nt)*32 + lane)*16 + 2r + pair], value = W[nt*16+lane%16][kt*32+kbase(r,lane/16)+pair]
__global__ __launch_bounds__(256) void wswz_kernel(const float* __restrict__ src,
                                                   bf16_t* __restrict__ dst,
                                                   int IN, int nnt) {
    int t = blockIdx.x * 256 + threadIdx.x;
    int total = (IN >> 5) * nnt * 256;
    if (t >= total) return;
    int r    = t & 7;
    int lane = (t >> 3) & 31;
    int rest = t >> 8;
    int nt   = rest % nnt;
    int kt   = rest / nnt;
    int k    = kt * 32 + kbase(r, lane >> 4);
    int n0   = nt * 16 + (lane & 15);
    dst[2 * t]     = (bf16_t)src[n0 * IN + k];
    dst[2 * t + 1] = (bf16_t)src[n0 * IN + k + 1];
}

// --------------------------------------------------------------------------
__global__ __launch_bounds__(256) void zero_kernel(float* __restrict__ p, size_t n) {
    size_t i = (size_t)blockIdx.x * 256 + threadIdx.x;
    if (i < n) p[i] = 0.0f;
}

// --------------------------------------------------------------------------
// prep1: X [N,F,3,3] -> compact decomposition, written in A-fragment-swizzled
// order: comp0[((tile*10 + c)*2 + kt)*512 + (m + 16 g)*16 + 2r + pair]
// channels: 0=lam, 1..3 = a01,a02,a12, 4..9 = s00,s01,s02,s11,s12,s22
__global__ __launch_bounds__(256) void prep1_kernel(const float* __restrict__ X,
                                                    bf16_t* __restrict__ comp0) {
    size_t gid = (size_t)blockIdx.x * 256 + threadIdx.x;
    int n = (int)(gid >> 6);
    int f = (int)(gid & 63);
    if (n >= N_ATOMS) return;
    const float* x = X + ((size_t)n * FDIM + f) * 9;
    float t[9];
    float nrm = 0.0f;
#pragma unroll
    for (int i = 0; i < 9; ++i) { t[i] = x[i]; nrm += t[i] * t[i]; }
    float inv = 1.0f / (nrm + 1.0f);
#pragma unroll
    for (int i = 0; i < 9; ++i) t[i] *= inv;
    float lam = (t[0] + t[4] + t[8]) * (1.0f / 3.0f);
    float v[10];
    v[0] = lam;
    v[1] = 0.5f * (t[1] - t[3]);
    v[2] = 0.5f * (t[2] - t[6]);
    v[3] = 0.5f * (t[5] - t[7]);
    v[4] = t[0] - lam;
    v[5] = 0.5f * (t[1] + t[3]);
    v[6] = 0.5f * (t[2] + t[6]);
    v[7] = t[4] - lam;
    v[8] = 0.5f * (t[5] + t[7]);
    v[9] = t[8] - lam;
    int tile = n >> 4, m = n & 15;
    int kt = f >> 5, k = f & 31, g, r, pair;
    kinv(k, g, r, pair);
    size_t base = ((size_t)tile * 10 * 2 + kt) * 512 + (m + 16 * g) * 16 + 2 * r + pair;
#pragma unroll
    for (int c = 0; c < 10; ++c) comp0[base + (size_t)c * 1024] = (bf16_t)v[c];
}

// --------------------------------------------------------------------------
// lin: per-channel feature linear via WMMA. Fully padded: no guards.
// in:  A-fragment-swizzled bf16  [N_TILES,10,2,512]
// out: normal layout f32 [N_PAD,10,F]
__global__ __launch_bounds__(256) void lin_kernel(const bf16_t* __restrict__ in,
                                                  float* __restrict__ out,
                                                  const bf16_t* __restrict__ wI,
                                                  const bf16_t* __restrict__ wA,
                                                  const bf16_t* __restrict__ wS) {
    const int lane = threadIdx.x & 31;
    const int wv   = threadIdx.x >> 5;
    const int wg   = blockIdx.x * 8 + wv;
    if (wg >= N_TILES * 10) return;  // wave-uniform
    const int tile = wg / 10;
    const int c    = wg % 10;
    const bf16_t* W = (c == 0) ? wI : ((c < 4) ? wA : wS);
    const int m = lane & 15, g = lane >> 4;

    const bf16_t* abase = in + ((size_t)(tile * 10 + c) * 2) * 512 + lane * 16;
    v16bf a[2];
#pragma unroll
    for (int kt = 0; kt < 2; ++kt) a[kt] = *(const v16bf*)(abase + kt * 512);

#pragma unroll
    for (int nt = 0; nt < 4; ++nt) {
        v8f acc = {0.f, 0.f, 0.f, 0.f, 0.f, 0.f, 0.f, 0.f};
#pragma unroll
        for (int kt = 0; kt < 2; ++kt) {
            v16bf b = *(const v16bf*)(W + (size_t)((kt * 4 + nt) * 32 + lane) * 16);
            acc = wmma_bf16(a[kt], b, acc);
        }
        float* obase = out + ((size_t)(tile * 16 + 8 * g) * 10 + c) * 64 + nt * 16 + m;
#pragma unroll
        for (int r = 0; r < 8; ++r) obase[(size_t)r * 640] = acc[r];
    }
}

// --------------------------------------------------------------------------
// edge MLP: rbf[E,32] -> 64 -> 128 -> 192, SiLU each layer, * cutoff.
// One wave per 16-edge tile. All MLP weights (68KB bf16, pre-swizzled) staged
// in LDS once per block; activation staging buffer kept in A-fragment order.
// Output layout [E][3][F] (coalesced reads in message kernel).
__global__ __launch_bounds__(256) void edge_mlp_kernel(
    const float* __restrict__ rfv, const float* __restrict__ dij,
    const bf16_t* __restrict__ wall,  // w1 | w2 | w3 contiguous, frag-swizzled
    const float* __restrict__ b1, const float* __restrict__ b2,
    const float* __restrict__ b3, bf16_t* __restrict__ hout) {
    __shared__ __align__(16) bf16_t wsh[2048 + 8192 + 24576];  // 68 KB
    __shared__ bf16_t hbuf[8][4 * 512];                        // 32 KB
    // cooperative copy of all weights into LDS: 17408 dwords = 4352 uint4
    {
        const uint4* s = (const uint4*)wall;
        uint4* d = (uint4*)wsh;
#pragma unroll
        for (int i = 0; i < 17; ++i) d[threadIdx.x + 256 * i] = s[threadIdx.x + 256 * i];
    }
    __syncthreads();
    const bf16_t* w1s = wsh;
    const bf16_t* w2s = wsh + 2048;
    const bf16_t* w3s = wsh + 10240;

    const int lane = threadIdx.x & 31;
    const int wv   = threadIdx.x >> 5;
    const int m = lane & 15, g = lane >> 4;
    const size_t e0 = (size_t)blockIdx.x * 128 + (size_t)wv * 16;  // E%128==0
    bf16_t* h = hbuf[wv];

    // ---------- layer 1: [16x32] x W1^T[32x64] ----------
    v16bf a1;
#pragma unroll
    for (int r = 0; r < 8; ++r) {
        int k = kbase(r, g);
        v2f v = *(const v2f*)(rfv + (e0 + m) * RBFDIM + k);
        a1[2 * r]     = (bf16_t)v.x;
        a1[2 * r + 1] = (bf16_t)v.y;
    }
#pragma unroll
    for (int nt = 0; nt < 4; ++nt) {
        v16bf b = *(const v16bf*)(w1s + (size_t)(nt * 32 + lane) * 16);
        v8f acc = {0.f, 0.f, 0.f, 0.f, 0.f, 0.f, 0.f, 0.f};
        acc = wmma_bf16(a1, b, acc);
        float bias = b1[nt * 16 + m];
        // D element (row = r+8g, col j = nt*16+m) -> A-swizzled LDS slot
        int j = nt * 16 + m;
        int ktj = j >> 5, kj = j & 31, gj, rj, pj;
        kinv(kj, gj, rj, pj);
        bf16_t* sb = h + ktj * 512 + 256 * gj + 2 * rj + pj;
#pragma unroll
        for (int r = 0; r < 8; ++r)
            sb[(r + 8 * g) * 16] = (bf16_t)silu_f(acc[r] + bias);
    }

    // ---------- layer 2: [16x64] x W2^T[64x128] ----------
    v16bf a2[2];  // preload h1 fragments (same-wave LDS ops are in-order)
#pragma unroll
    for (int kt = 0; kt < 2; ++kt)
        a2[kt] = *(const v16bf*)(h + kt * 512 + lane * 16);
#pragma unroll
    for (int nt = 0; nt < 8; ++nt) {
        v8f acc = {0.f, 0.f, 0.f, 0.f, 0.f, 0.f, 0.f, 0.f};
#pragma unroll
        for (int kt = 0; kt < 2; ++kt) {
            v16bf b = *(const v16bf*)(w2s + (size_t)((kt * 8 + nt) * 32 + lane) * 16);
            acc = wmma_bf16(a2[kt], b, acc);
        }
        float bias = b2[nt * 16 + m];
        int j = nt * 16 + m;
        int ktj = j >> 5, kj = j & 31, gj, rj, pj;
        kinv(kj, gj, rj, pj);
        bf16_t* sb = h + ktj * 512 + 256 * gj + 2 * rj + pj;
#pragma unroll
        for (int r = 0; r < 8; ++r)
            sb[(r + 8 * g) * 16] = (bf16_t)silu_f(acc[r] + bias);
    }

    // ---------- layer 3: [16x128] x W3^T[128x192], * cutoff ----------
    v16bf a3[4];
#pragma unroll
    for (int kt = 0; kt < 4; ++kt)
        a3[kt] = *(const v16bf*)(h + kt * 512 + lane * 16);
    float Cm[8];
#pragma unroll
    for (int r = 0; r < 8; ++r) {
        float d = dij[e0 + r + 8 * g];
        Cm[r] = (d < 5.0f) ? 0.5f * (__cosf(d * 0.6283185307f) + 1.0f) : 0.0f;
    }
#pragma unroll
    for (int nt = 0; nt < 12; ++nt) {
        v8f acc = {0.f, 0.f, 0.f, 0.f, 0.f, 0.f, 0.f, 0.f};
#pragma unroll
        for (int kt = 0; kt < 4; ++kt) {
            v16bf b = *(const v16bf*)(w3s + (size_t)((kt * 12 + nt) * 32 + lane) * 16);
            acc = wmma_bf16(a3[kt], b, acc);
        }
        float bias = b3[nt * 16 + m];
        // col j = nt*16+m maps to (f = j/3, comp = j%3); store layout [E][3][F]
        int j  = nt * 16 + m;
        int fj = j / 3, cj = j - 3 * fj;
        bf16_t* ob = hout + (e0 + 8 * g) * 192 + cj * 64 + fj;
#pragma unroll
        for (int r = 0; r < 8; ++r)
            ob[(size_t)r * 192] = (bf16_t)(silu_f(acc[r] + bias) * Cm[r]);
    }
}

// --------------------------------------------------------------------------
// message passing: gather compact comp at dst, scale by edge coefs, atomic
// scatter-add into msg[src]. 64 threads (features) per edge.
// coef layout [E][3][F] -> fully coalesced 2-byte loads.
__global__ __launch_bounds__(256) void message_kernel(const bf16_t* __restrict__ coef,
                                                      const float* __restrict__ comp,
                                                      const int* __restrict__ pair,
                                                      float* __restrict__ msg) {
    int e = blockIdx.x * 4 + (threadIdx.x >> 6);
    int f = threadIdx.x & 63;
    int src = pair[e];
    int dst = pair[EDGES + e];
    const bf16_t* cf = coef + (size_t)e * 192 + f;
    float c0 = (float)cf[0], c1 = (float)cf[64], c2 = (float)cf[128];
    const float* in  = comp + (size_t)dst * 640 + f;
    float*       acc = msg  + (size_t)src * 640 + f;
    __hip_atomic_fetch_add(acc, c0 * in[0], __ATOMIC_RELAXED, __HIP_MEMORY_SCOPE_AGENT);
#pragma unroll
    for (int c = 1; c < 4; ++c)
        __hip_atomic_fetch_add(acc + c * 64, c1 * in[c * 64], __ATOMIC_RELAXED,
                               __HIP_MEMORY_SCOPE_AGENT);
#pragma unroll
    for (int c = 4; c < 10; ++c)
        __hip_atomic_fetch_add(acc + c * 64, c2 * in[c * 64], __ATOMIC_RELAXED,
                               __HIP_MEMORY_SCOPE_AGENT);
}

// --------------------------------------------------------------------------
__device__ __forceinline__ void build33(float* M, const float* v) {
    float iso = v[0];
    M[0] = iso + v[4]; M[1] = v[5] + v[1]; M[2] = v[6] + v[2];
    M[3] = v[5] - v[1]; M[4] = iso + v[7]; M[5] = v[8] + v[3];
    M[6] = v[6] - v[2]; M[7] = v[8] - v[3]; M[8] = iso + v[9];
}
__device__ __forceinline__ void mm33(float* C, const float* A, const float* B) {
#pragma unroll
    for (int i = 0; i < 3; ++i)
#pragma unroll
        for (int j = 0; j < 3; ++j)
            C[i * 3 + j] = A[i * 3 + 0] * B[0 * 3 + j] + A[i * 3 + 1] * B[1 * 3 + j] +
                           A[i * 3 + 2] * B[2 * 3 + j];
}

// prep2: msg + Y -> normalized decomposition of scale*(M@Y + Y@M),
// written A-fragment-swizzled (bf16) for the second WMMA linear.
__global__ __launch_bounds__(256) void prep2_kernel(const float* __restrict__ msg,
                                                    const float* __restrict__ comp,
                                                    const float* __restrict__ q,
                                                    bf16_t* __restrict__ comp2) {
    size_t gid = (size_t)blockIdx.x * 256 + threadIdx.x;
    int n = (int)(gid >> 6);
    int f = (int)(gid & 63);
    if (n >= N_ATOMS) return;
    float mv[10], yv[10];
#pragma unroll
    for (int c = 0; c < 10; ++c) {
        mv[c] = msg[((size_t)n * 10 + c) * FDIM + f];
        yv[c] = comp[((size_t)n * 10 + c) * FDIM + f];
    }
    float M[9], Y[9], MA[9], MB[9], T[9];
    build33(M, mv);
    build33(Y, yv);
    mm33(MA, M, Y);
    mm33(MB, Y, M);
    float scale = 1.0f + 0.1f * q[n];
    float nrm = 0.0f;
#pragma unroll
    for (int i = 0; i < 9; ++i) { T[i] = scale * (MA[i] + MB[i]); nrm += T[i] * T[i]; }
    float inv = 1.0f / (nrm + 1.0f);
    float lam = (T[0] + T[4] + T[8]) * (1.0f / 3.0f);
    float v[10];
    v[0] = lam;
    v[1] = 0.5f * (T[1] - T[3]);
    v[2] = 0.5f * (T[2] - T[6]);
    v[3] = 0.5f * (T[5] - T[7]);
    v[4] = T[0] - lam;
    v[5] = 0.5f * (T[1] + T[3]);
    v[6] = 0.5f * (T[2] + T[6]);
    v[7] = T[4] - lam;
    v[8] = 0.5f * (T[5] + T[7]);
    v[9] = T[8] - lam;
    int tile = n >> 4, m = n & 15;
    int kt = f >> 5, k = f & 31, g, r, pair;
    kinv(k, g, r, pair);
    size_t base = ((size_t)tile * 10 * 2 + kt) * 512 + (m + 16 * g) * 16 + 2 * r + pair;
#pragma unroll
    for (int c = 0; c < 10; ++c) comp2[base + (size_t)c * 1024] = (bf16_t)(v[c] * inv);
}

// --------------------------------------------------------------------------
// final: out = Xn + dX + scale * (dX @ dX)
__global__ __launch_bounds__(256) void final_kernel(const float* __restrict__ X,
                                                    const float* __restrict__ dx,
                                                    const float* __restrict__ q,
                                                    float* __restrict__ out) {
    size_t gid = (size_t)blockIdx.x * 256 + threadIdx.x;
    int n = (int)(gid >> 6);
    int f = (int)(gid & 63);
    if (n >= N_ATOMS) return;
    const float* x = X + ((size_t)n * FDIM + f) * 9;
    float xn[9];
    float nrm = 0.0f;
#pragma unroll
    for (int i = 0; i < 9; ++i) { xn[i] = x[i]; nrm += xn[i] * xn[i]; }
    float inv = 1.0f / (nrm + 1.0f);
#pragma unroll
    for (int i = 0; i < 9; ++i) xn[i] *= inv;
    float v[10];
#pragma unroll
    for (int c = 0; c < 10; ++c) v[c] = dx[((size_t)n * 10 + c) * FDIM + f];
    float D[9], P[9];
    build33(D, v);
    mm33(P, D, D);
    float scale = 1.0f + 0.1f * q[n];
    float* o = out + ((size_t)n * FDIM + f) * 9;
#pragma unroll
    for (int i = 0; i < 9; ++i) o[i] = xn[i] + D[i] + scale * P[i];
}

// ==========================================================================
extern "C" void kernel_launch(void* const* d_in, const int* in_sizes, int n_in,
                              void* d_out, int out_size, void* d_ws, size_t ws_size,
                              hipStream_t stream) {
    const float* X    = (const float*)d_in[0];
    const int*   pair = (const int*)d_in[1];
    const float* dij  = (const float*)d_in[2];
    const float* rfv  = (const float*)d_in[3];
    const float* q    = (const float*)d_in[4];
    const float* W1   = (const float*)d_in[5];
    const float* b1   = (const float*)d_in[6];
    const float* W2   = (const float*)d_in[7];
    const float* b2   = (const float*)d_in[8];
    const float* W3   = (const float*)d_in[9];
    const float* b3   = (const float*)d_in[10];
    const float* Wl[6];
    for (int i = 0; i < 6; ++i) Wl[i] = (const float*)d_in[11 + i];

    char* ws = (char*)d_ws;
    // B-fragment-swizzled bf16 weights at base of workspace (w1|w2|w3 contiguous)
    bf16_t* w1bf = (bf16_t*)ws;          // 2048
    bf16_t* w2bf = w1bf + 64 * 32;       // 8192
    bf16_t* w3bf = w2bf + 128 * 64;      // 24576
    bf16_t* wlbf = w3bf + 192 * 128;     // 6 * 4096
    // large regions (sequentially reused)
    bf16_t* comp0 = (bf16_t*)(ws + (1ull << 20));            // [N_TILES,10,2,512] bf16
    float*  comp  = (float*)(ws + 33ull * 1024 * 1024);      // [N_PAD,10,F] f32
    float*  msg   = (float*)(ws + 97ull * 1024 * 1024);      // [N_PAD,10,F] f32
    bf16_t* rfvbf = (bf16_t*)(ws + 161ull * 1024 * 1024);    // [E,3,F] bf16
    bf16_t* comp2 = comp0;          // reuse: comp0 consumed before prep2
    float*  dx    = (float*)rfvbf;  // reuse: rfv coefs consumed before lin2

    // 1) pre-swizzle weights to WMMA B-fragment order (bf16)
    wswz_kernel<<<4, 256, 0, stream>>>(W1, w1bf, 32, 4);     // 64x32
    wswz_kernel<<<16, 256, 0, stream>>>(W2, w2bf, 64, 8);    // 128x64
    wswz_kernel<<<48, 256, 0, stream>>>(W3, w3bf, 128, 12);  // 192x128
    for (int i = 0; i < 6; ++i)
        wswz_kernel<<<8, 256, 0, stream>>>(Wl[i], wlbf + i * 4096, 64, 4);  // 64x64

    // 2) zero message accumulator (padded: N_PAD*640 floats)
    zero_kernel<<<62520, 256, 0, stream>>>(msg, (size_t)N_PAD * 640);

    // 3) decompose X -> compact bf16 components (A-fragment-swizzled)
    prep1_kernel<<<6250, 256, 0, stream>>>(X, comp0);

    // 4) feature-linear (Wl0/Wl1/Wl2) via WMMA
    lin_kernel<<<1954, 256, 0, stream>>>(comp0, comp, wlbf, wlbf + 4096, wlbf + 2 * 4096);

    // 5) edge MLP via WMMA (E/128 = 3125 blocks), weights staged in LDS
    edge_mlp_kernel<<<3125, 256, 0, stream>>>(rfv, dij, w1bf, b1, b2, b3, rfvbf);

    // 6) message passing (E*F threads)
    message_kernel<<<100000, 256, 0, stream>>>(rfvbf, comp, pair, msg);

    // 7) O(3) branch elementwise -> swizzled comp2
    prep2_kernel<<<6250, 256, 0, stream>>>(msg, comp, q, comp2);

    // 8) feature-linear (Wl3/Wl4/Wl5) via WMMA
    lin_kernel<<<1954, 256, 0, stream>>>(comp2, dx, wlbf + 3 * 4096, wlbf + 4 * 4096,
                                         wlbf + 5 * 4096);

    // 9) final reconstruction
    final_kernel<<<6250, 256, 0, stream>>>(X, dx, q, (float*)d_out);
}
